// MultiLayer_BTreeLSTM_83099027243629
// MI455X (gfx1250) — compile-verified, matrix operand order checked by reading, not run on hardware
//
#include <hip/hip_runtime.h>
#include <hip/hip_bf16.h>
#include <stdint.h>

// ---------------- problem constants (match setup_inputs) ----------------
#define B_   256
#define N_   256
#define IN_  512
#define H_   256
#define L_   2
#define G5   (5 * H_)          // 1280 fwd gate dim
#define G4   (4 * H_)          // 1024 bwd gate dim
#define KF   (IN_ + 2 * H_)    // 1024 fwd reduction dim  [x | lh | rh]
#define KB   (IN_ + H_)        // 768  bwd reduction dim  [x | ph]
#define APITCH_F (KF + 8)      // LDS row pitch (halves): 16B-aligned rows, bank-staggered
#define APITCH_B (KB + 8)
#define MTILE 32               // rows of A staged per block (2 x 16-row WMMA sub-tiles)

typedef __attribute__((ext_vector_type(16))) __bf16 v16bf;
typedef __attribute__((ext_vector_type(8)))  float  v8f;

union FragB { v16bf v; uint4 u[2]; };

__device__ __forceinline__ unsigned short f2bf(float f) {
  unsigned int u = __float_as_uint(f);
  u += 0x7FFFu + ((u >> 16) & 1u);          // round-to-nearest-even
  return (unsigned short)(u >> 16);
}
__device__ __forceinline__ float sigf(float x) { return 1.0f / (1.0f + __expf(-x)); }
__device__ __forceinline__ float tanh_f(float x) { return 2.0f * sigf(2.0f * x) - 1.0f; }

// ---------------- weight packing: fp32 -> bf16, K-concatenated ----------------
// Wf[l]: (1280 x 1024) = [fwd_Wx(512) | fwd_Wl(256) | fwd_Wr(256)]
__global__ void pack_fwd_w(const float* __restrict__ Wx, const float* __restrict__ Wl,
                           const float* __restrict__ Wr, unsigned short* __restrict__ out) {
  const int total = L_ * G5 * KF;
  for (int i = blockIdx.x * blockDim.x + threadIdx.x; i < total; i += gridDim.x * blockDim.x) {
    const int l   = i / (G5 * KF);
    const int rem = i - l * (G5 * KF);
    const int g   = rem / KF;
    const int k   = rem - g * KF;
    float v;
    if (k < IN_)           v = Wx[((size_t)l * G5 + g) * IN_ + k];
    else if (k < IN_ + H_) v = Wl[((size_t)l * G5 + g) * H_ + (k - IN_)];
    else                   v = Wr[((size_t)l * G5 + g) * H_ + (k - IN_ - H_)];
    out[i] = f2bf(v);
  }
}
// Wb[l]: (1024 x 768) = [bwd_Wx(512) | bwd_Wh(256)]
__global__ void pack_bwd_w(const float* __restrict__ Wx, const float* __restrict__ Wh,
                           unsigned short* __restrict__ out) {
  const int total = L_ * G4 * KB;
  for (int i = blockIdx.x * blockDim.x + threadIdx.x; i < total; i += gridDim.x * blockDim.x) {
    const int l   = i / (G4 * KB);
    const int rem = i - l * (G4 * KB);
    const int g   = rem / KB;
    const int k   = rem - g * KB;
    float v;
    if (k < IN_) v = Wx[((size_t)l * G4 + g) * IN_ + k];
    else         v = Wh[((size_t)l * G4 + g) * H_ + (k - IN_)];
    out[i] = f2bf(v);
  }
}

// ---------------- fused forward (leaves-to-root) level kernel ----------------
// rows r in [0, B*cnt): b = r>>lcShift, node = start + (r & (cnt-1))
// A row (bf16, K=1024): [ x[b,node,0:256] | x[b,node,256:512] | ml*h[left] | mr*h[right] ]
__global__ __launch_bounds__(128)
void fwd_level_kernel(const float* __restrict__ xf32,            // layer0 features (or null)
                      const unsigned short* __restrict__ xa_bf,  // prev-layer fwd h (or null)
                      const unsigned short* __restrict__ xb_bf,  // prev-layer bwd h (or null)
                      const unsigned short* __restrict__ W,      // (1280 x 1024) bf16
                      const float* __restrict__ bx, const float* __restrict__ bl,
                      const float* __restrict__ br,
                      float* __restrict__ cbuf,                  // (B,N,H) f32
                      unsigned short* __restrict__ hbuf,         // (B,N,H) bf16
                      float* __restrict__ outp,                  // d_out (last layer) or null
                      int start, int lcShift) {
  __shared__ __align__(16) unsigned short As[MTILE * APITCH_F];
  const int tid  = threadIdx.x;
  const int cnt  = 1 << lcShift;
  const int rowB = blockIdx.x * MTILE;

  // ---- cooperative A-tile staging (32 rows x 1024 halves) ----
  for (int i = tid; i < MTILE * (KF / 8); i += 128) {
    const int rr = i / (KF / 8);
    const int k0 = (i % (KF / 8)) * 8;
    const int r  = rowB + rr;
    const int b  = r >> lcShift;
    const int node = start + (r & (cnt - 1));
    const int seg = k0 >> 8;          // 4 segments of 256 halves
    const int ks  = k0 & 255;
    uint4 val; val.x = val.y = val.z = val.w = 0u;
    if (seg <= 1) {
      if (xf32) {  // layer 0: fp32 features, convert on the fly
        const float* p = xf32 + ((size_t)(b * N_ + node) * IN_) + seg * 256 + ks;
        val.x = (unsigned)f2bf(p[0]) | ((unsigned)f2bf(p[1]) << 16);
        val.y = (unsigned)f2bf(p[2]) | ((unsigned)f2bf(p[3]) << 16);
        val.z = (unsigned)f2bf(p[4]) | ((unsigned)f2bf(p[5]) << 16);
        val.w = (unsigned)f2bf(p[6]) | ((unsigned)f2bf(p[7]) << 16);
      } else {     // later layers: x = concat(fwd_h, bwd_h) already bf16
        const unsigned short* p = (seg == 0 ? xa_bf : xb_bf) + (size_t)(b * N_ + node) * H_ + ks;
        val = *(const uint4*)p;
      }
    } else {       // child hidden states (deeper level already computed); masked to 0
      const int child = 2 * node + (seg == 2 ? 1 : 2);
      if (child < N_) val = *(const uint4*)(hbuf + (size_t)(b * N_ + child) * H_ + ks);
    }
    *(uint4*)&As[rr * APITCH_F + k0] = val;
  }
  __syncthreads();

  const int wave  = tid >> 5;
  const int lane  = tid & 31;
  const int m     = lane & 15;   // A row / B column within tile
  const int hi    = lane >> 4;   // half-wave K-striping select
  const int hBase = blockIdx.y * 64 + wave * 16;

  v8f acc[2][5];                 // 2 row sub-tiles x 5 gates
  #pragma unroll
  for (int t = 0; t < 2; ++t)
    #pragma unroll
    for (int g = 0; g < 5; ++g)
      #pragma unroll
      for (int v = 0; v < 8; ++v) acc[t][g][v] = 0.0f;

  #pragma unroll 2
  for (int kk = 0; kk < KF; kk += 32) {
    FragB a0, a1;
    a0.u[0] = *(const uint4*)&As[m * APITCH_F + kk + hi * 8];
    a0.u[1] = *(const uint4*)&As[m * APITCH_F + kk + 16 + hi * 8];
    a1.u[0] = *(const uint4*)&As[(m + 16) * APITCH_F + kk + hi * 8];
    a1.u[1] = *(const uint4*)&As[(m + 16) * APITCH_F + kk + 16 + hi * 8];
    #pragma unroll
    for (int g = 0; g < 5; ++g) {
      const unsigned short* wp = W + (size_t)(g * H_ + hBase + m) * KF + kk + hi * 8;
      FragB bfr;
      bfr.u[0] = *(const uint4*)wp;
      bfr.u[1] = *(const uint4*)(wp + 16);
      acc[0][g] = __builtin_amdgcn_wmma_f32_16x16x32_bf16(false, a0.v, false, bfr.v,
                                                          (short)0, acc[0][g], false, false);
      acc[1][g] = __builtin_amdgcn_wmma_f32_16x16x32_bf16(false, a1.v, false, bfr.v,
                                                          (short)0, acc[1][g], false, false);
    }
    if (kk + 32 < KF)  // L2 hint for next K panel of weights
      __builtin_prefetch(W + (size_t)(hBase + m) * KF + kk + 32, 0, 3);
  }

  // ---- LSTM cell epilogue ----
  const int n = lane & 15;
  const int h = hBase + n;
  float bI[5], bL[5], bR[5];
  #pragma unroll
  for (int g = 0; g < 5; ++g) {
    bI[g] = bx[g * H_ + h]; bL[g] = bl[g * H_ + h]; bR[g] = br[g * H_ + h];
  }
  #pragma unroll
  for (int t = 0; t < 2; ++t) {
    #pragma unroll
    for (int v = 0; v < 8; ++v) {
      const int r = rowB + t * 16 + v + hi * 8;
      const int b = r >> lcShift;
      const int node = start + (r & (cnt - 1));
      const int li = 2 * node + 1, ri = 2 * node + 2;
      const bool ml = li < N_, mr = ri < N_;
      const float lc = ml ? cbuf[(size_t)(b * N_ + li) * H_ + h] : 0.0f;
      const float rc = mr ? cbuf[(size_t)(b * N_ + ri) * H_ + h] : 0.0f;
      const float gi  = acc[t][0][v] + bI[0] + (ml ? bL[0] : 0.f) + (mr ? bR[0] : 0.f);
      const float go  = acc[t][1][v] + bI[1] + (ml ? bL[1] : 0.f) + (mr ? bR[1] : 0.f);
      const float gfl = acc[t][2][v] + bI[2] + (ml ? bL[2] : 0.f) + (mr ? bR[2] : 0.f);
      const float gfr = acc[t][3][v] + bI[3] + (ml ? bL[3] : 0.f) + (mr ? bR[3] : 0.f);
      const float gu  = acc[t][4][v] + bI[4] + (ml ? bL[4] : 0.f) + (mr ? bR[4] : 0.f);
      const float cn = sigf(gi) * tanh_f(gu) + sigf(gfl) * lc + sigf(gfr) * rc;
      const float hn = sigf(go) * tanh_f(cn);
      const size_t o = (size_t)(b * N_ + node) * H_ + h;
      cbuf[o] = cn;
      hbuf[o] = f2bf(hn);
      if (outp) outp[(size_t)(b * N_ + node) * (2 * H_) + h] = hn;   // fwd half of concat
    }
  }
}

// ---------------- fused backward (root-to-leaves) level kernel ----------------
// A row (bf16, K=768): [ x[b,node,0:256] | x[b,node,256:512] | h[parent] ]
__global__ __launch_bounds__(128)
void bwd_level_kernel(const float* __restrict__ xf32,
                      const unsigned short* __restrict__ xa_bf,
                      const unsigned short* __restrict__ xb_bf,
                      const unsigned short* __restrict__ W,      // (1024 x 768) bf16
                      const float* __restrict__ bx, const float* __restrict__ bh,
                      float* __restrict__ cbuf,
                      unsigned short* __restrict__ hbuf,
                      float* __restrict__ outp,
                      int start, int lcShift) {
  __shared__ __align__(16) unsigned short As[MTILE * APITCH_B];
  const int tid  = threadIdx.x;
  const int cnt  = 1 << lcShift;
  const int rowB = blockIdx.x * MTILE;

  for (int i = tid; i < MTILE * (KB / 8); i += 128) {
    const int rr = i / (KB / 8);
    const int k0 = (i % (KB / 8)) * 8;
    const int r  = rowB + rr;
    const int b  = r >> lcShift;
    const int node = start + (r & (cnt - 1));
    const int seg = k0 >> 8;
    const int ks  = k0 & 255;
    uint4 val; val.x = val.y = val.z = val.w = 0u;
    if (seg <= 1) {
      if (xf32) {
        const float* p = xf32 + ((size_t)(b * N_ + node) * IN_) + seg * 256 + ks;
        val.x = (unsigned)f2bf(p[0]) | ((unsigned)f2bf(p[1]) << 16);
        val.y = (unsigned)f2bf(p[2]) | ((unsigned)f2bf(p[3]) << 16);
        val.z = (unsigned)f2bf(p[4]) | ((unsigned)f2bf(p[5]) << 16);
        val.w = (unsigned)f2bf(p[6]) | ((unsigned)f2bf(p[7]) << 16);
      } else {
        const unsigned short* p = (seg == 0 ? xa_bf : xb_bf) + (size_t)(b * N_ + node) * H_ + ks;
        val = *(const uint4*)p;
      }
    } else if (node > 0) {   // parent hidden (shallower level, already computed)
      const int par = (node - 1) >> 1;
      val = *(const uint4*)(hbuf + (size_t)(b * N_ + par) * H_ + ks);
    }
    *(uint4*)&As[rr * APITCH_B + k0] = val;
  }
  __syncthreads();

  const int wave  = tid >> 5;
  const int lane  = tid & 31;
  const int m     = lane & 15;
  const int hi    = lane >> 4;
  const int hBase = blockIdx.y * 64 + wave * 16;

  v8f acc[2][4];
  #pragma unroll
  for (int t = 0; t < 2; ++t)
    #pragma unroll
    for (int g = 0; g < 4; ++g)
      #pragma unroll
      for (int v = 0; v < 8; ++v) acc[t][g][v] = 0.0f;

  #pragma unroll 2
  for (int kk = 0; kk < KB; kk += 32) {
    FragB a0, a1;
    a0.u[0] = *(const uint4*)&As[m * APITCH_B + kk + hi * 8];
    a0.u[1] = *(const uint4*)&As[m * APITCH_B + kk + 16 + hi * 8];
    a1.u[0] = *(const uint4*)&As[(m + 16) * APITCH_B + kk + hi * 8];
    a1.u[1] = *(const uint4*)&As[(m + 16) * APITCH_B + kk + 16 + hi * 8];
    #pragma unroll
    for (int g = 0; g < 4; ++g) {
      const unsigned short* wp = W + (size_t)(g * H_ + hBase + m) * KB + kk + hi * 8;
      FragB bfr;
      bfr.u[0] = *(const uint4*)wp;
      bfr.u[1] = *(const uint4*)(wp + 16);
      acc[0][g] = __builtin_amdgcn_wmma_f32_16x16x32_bf16(false, a0.v, false, bfr.v,
                                                          (short)0, acc[0][g], false, false);
      acc[1][g] = __builtin_amdgcn_wmma_f32_16x16x32_bf16(false, a1.v, false, bfr.v,
                                                          (short)0, acc[1][g], false, false);
    }
    if (kk + 32 < KB)
      __builtin_prefetch(W + (size_t)(hBase + m) * KB + kk + 32, 0, 3);
  }

  const int n = lane & 15;
  const int h = hBase + n;
  float bC[4];
  #pragma unroll
  for (int g = 0; g < 4; ++g) bC[g] = bx[g * H_ + h] + bh[g * H_ + h];

  #pragma unroll
  for (int t = 0; t < 2; ++t) {
    #pragma unroll
    for (int v = 0; v < 8; ++v) {
      const int r = rowB + t * 16 + v + hi * 8;
      const int b = r >> lcShift;
      const int node = start + (r & (cnt - 1));
      const float pc = (node > 0) ? cbuf[(size_t)(b * N_ + ((node - 1) >> 1)) * H_ + h] : 0.0f;
      const float gi = acc[t][0][v] + bC[0];
      const float go = acc[t][1][v] + bC[1];
      const float gf = acc[t][2][v] + bC[2];
      const float gu = acc[t][3][v] + bC[3];
      const float cn = sigf(gi) * tanh_f(gu) + sigf(gf) * pc;
      const float hn = sigf(go) * tanh_f(cn);
      const size_t o = (size_t)(b * N_ + node) * H_ + h;
      cbuf[o] = cn;
      hbuf[o] = f2bf(hn);
      if (outp) outp[(size_t)(b * N_ + node) * (2 * H_) + H_ + h] = hn;  // bwd half of concat
    }
  }
}

// ---------------- host orchestration ----------------
static inline size_t alignup(size_t x) { return (x + 255) & ~(size_t)255; }

extern "C" void kernel_launch(void* const* d_in, const int* in_sizes, int n_in,
                              void* d_out, int out_size, void* d_ws, size_t ws_size,
                              hipStream_t stream) {
  const float* features = (const float*)d_in[0];
  const float* fwd_Wx   = (const float*)d_in[1];
  const float* fwd_bx   = (const float*)d_in[2];
  const float* fwd_Wl   = (const float*)d_in[3];
  const float* fwd_bl   = (const float*)d_in[4];
  const float* fwd_Wr   = (const float*)d_in[5];
  const float* fwd_br   = (const float*)d_in[6];
  const float* bwd_Wx   = (const float*)d_in[7];
  const float* bwd_bx   = (const float*)d_in[8];
  const float* bwd_Wh   = (const float*)d_in[9];
  const float* bwd_bh   = (const float*)d_in[10];

  char* ws = (char*)d_ws;
  size_t off = 0;
  unsigned short* Wf = (unsigned short*)(ws + off); off = alignup(off + (size_t)L_ * G5 * KF * 2);
  unsigned short* Wb = (unsigned short*)(ws + off); off = alignup(off + (size_t)L_ * G4 * KB * 2);
  unsigned short* fwdH[2]; unsigned short* bwdH[2];
  for (int i = 0; i < 2; ++i) { fwdH[i] = (unsigned short*)(ws + off); off = alignup(off + (size_t)B_ * N_ * H_ * 2); }
  for (int i = 0; i < 2; ++i) { bwdH[i] = (unsigned short*)(ws + off); off = alignup(off + (size_t)B_ * N_ * H_ * 2); }
  float* cbuf = (float*)(ws + off); off = alignup(off + (size_t)B_ * N_ * H_ * 4);
  if (off > ws_size) return;  // workspace too small; nothing sane to do

  pack_fwd_w<<<512, 256, 0, stream>>>(fwd_Wx, fwd_Wl, fwd_Wr, Wf);
  pack_bwd_w<<<512, 256, 0, stream>>>(bwd_Wx, bwd_Wh, Wb);

  for (int l = 0; l < L_; ++l) {
    const float*          xf = (l == 0) ? features : nullptr;
    const unsigned short* xa = (l == 0) ? nullptr : fwdH[(l - 1) & 1];
    const unsigned short* xb = (l == 0) ? nullptr : bwdH[(l - 1) & 1];
    unsigned short* fh = fwdH[l & 1];
    unsigned short* bh = bwdH[l & 1];
    float* outp = (l == L_ - 1) ? (float*)d_out : nullptr;
    const unsigned short* Wfl = Wf + (size_t)l * G5 * KF;
    const unsigned short* Wbl = Wb + (size_t)l * G4 * KB;

    // forward: leaves -> root (deepest level first)
    for (int d = 8; d >= 0; --d) {
      const int startN = (1 << d) - 1;
      int cntN = (1 << d); if (cntN > N_ - startN) cntN = N_ - startN;
      int sh = 0; while ((1 << sh) < cntN) ++sh;
      dim3 grid((B_ * cntN) / MTILE, H_ / 64);
      fwd_level_kernel<<<grid, 128, 0, stream>>>(xf, xa, xb, Wfl,
          fwd_bx + (size_t)l * G5, fwd_bl + (size_t)l * G5, fwd_br + (size_t)l * G5,
          cbuf, fh, outp, startN, sh);
    }
    // backward: root -> leaves
    for (int d = 0; d <= 8; ++d) {
      const int startN = (1 << d) - 1;
      int cntN = (1 << d); if (cntN > N_ - startN) cntN = N_ - startN;
      int sh = 0; while ((1 << sh) < cntN) ++sh;
      dim3 grid((B_ * cntN) / MTILE, H_ / 64);
      bwd_level_kernel<<<grid, 128, 0, stream>>>(xf, xa, xb, Wbl,
          bwd_bx + (size_t)l * G4, bwd_bh + (size_t)l * G4,
          cbuf, bh, outp, startN, sh);
    }
  }
}
